// SpectralKnowledgeBase_48954037240518
// MI455X (gfx1250) — matrix operand order repaired and checked by reading, not compiled, not run on hardware
//
#include <hip/hip_runtime.h>
#include <hip/hip_bf16.h>
#include <cmath>

// ---------------------------------------------------------------------------
// SpectralKnowledgeBase for MI455X (gfx1250, wave32, WMMA).
// GEMMs on v_wmma_f32_16x16x32_bf16 (bf16 operands, f32 accumulate).
// Round 3: A-tile staging via GLOBAL_LOAD_ASYNC_TO_LDS_B128 (ASYNCcnt path),
// B-tile transpose staged as packed k-pair b32 LDS stores.
// ---------------------------------------------------------------------------

typedef __bf16 bf16_t;
typedef __attribute__((ext_vector_type(16))) __bf16 v16bf;
typedef __attribute__((ext_vector_type(8)))  __bf16 v8bf;
typedef __attribute__((ext_vector_type(2)))  __bf16 v2bf;
typedef __attribute__((ext_vector_type(8)))  float  v8f;

#define EPSF 1e-8f

// Problem-instance constants (B=4, S=512, D=768, N=1000, V=50000, k=5)
constexpr int MROWS = 2048;
constexpr int DDIM  = 768;
constexpr int NCON  = 1000;
constexpr int NPAD  = 1024;   // padded concept count for tiling
constexpr int VOUT  = 50000;
constexpr int TOPK  = 5;

__device__ __forceinline__ float gelu_exact(float x) {
    return 0.5f * x * (1.0f + erff(x * 0.70710678118654752440f));
}

// global -> LDS async copy of 16 bytes (per lane), tracked by ASYNCcnt
__device__ __forceinline__ void async_copy_b128(uint32_t lds_byte_addr, const void* gptr) {
    asm volatile("global_load_async_to_lds_b128 %0, %1, off"
                 :: "v"(lds_byte_addr), "v"((unsigned long long)(uintptr_t)gptr)
                 : "memory");
}
__device__ __forceinline__ void wait_async0() {
    asm volatile("s_wait_asynccnt 0x0" ::: "memory");
}

// ---------------------------------------------------------------------------
// prep: magnitudes (bf16), inverse row norms, bf16 concat(real,imag)
// ---------------------------------------------------------------------------
__global__ void prep_mag_kernel(const float* __restrict__ sr, const float* __restrict__ si,
                                bf16_t* __restrict__ magbf, float* __restrict__ inv_mnorm,
                                bf16_t* __restrict__ xcat)
{
    __shared__ float red[256];
    const int row = blockIdx.x;
    const int tid = threadIdx.x;
    float s = 0.f;
    for (int d = tid; d < DDIM; d += 256) {
        float r = sr[(size_t)row * DDIM + d];
        float i = si[(size_t)row * DDIM + d];
        float m = sqrtf(r * r + i * i);
        magbf[(size_t)row * DDIM + d] = (bf16_t)m;
        xcat[(size_t)row * (2 * DDIM) + d]        = (bf16_t)r;
        xcat[(size_t)row * (2 * DDIM) + DDIM + d] = (bf16_t)i;
        s += m * m;
    }
    red[tid] = s;
    __syncthreads();
    for (int st = 128; st > 0; st >>= 1) {
        if (tid < st) red[tid] += red[tid + st];
        __syncthreads();
    }
    if (tid == 0) inv_mnorm[row] = 1.0f / fmaxf(sqrtf(red[0]), EPSF);
}

// concept magnitudes, stored transposed as cmagT[d][NPAD]; pad cols zeroed
__global__ void prep_concept_kernel(const float* __restrict__ cr, const float* __restrict__ ci,
                                    bf16_t* __restrict__ cmagT, float* __restrict__ inv_cnorm)
{
    __shared__ float red[256];
    const int n   = blockIdx.x;     // 0..NPAD-1
    const int tid = threadIdx.x;
    const bool live = (n < NCON);
    float s = 0.f;
    for (int d = tid; d < DDIM; d += 256) {
        float c = 0.f;
        if (live) {
            float r = cr[(size_t)n * DDIM + d];
            float i = ci[(size_t)n * DDIM + d];
            c = sqrtf(r * r + i * i);
        }
        cmagT[(size_t)d * NPAD + n] = (bf16_t)c;
        s += c * c;
    }
    red[tid] = s;
    __syncthreads();
    for (int st = 128; st > 0; st >>= 1) {
        if (tid < st) red[tid] += red[tid + st];
        __syncthreads();
    }
    if (tid == 0) inv_cnorm[n] = live ? (1.0f / fmaxf(sqrtf(red[0]), EPSF)) : 0.0f;
}

// ---------------------------------------------------------------------------
// WMMA fragment helpers (CDNA5 documented VGPR layouts, wave32)
// A 16x32 bf16: lane m=lane&15, half h=lane>>4; elt j -> K = (j<8?0:16)+h*8+(j&7)
// B 32x16 bf16 (staged transposed in LDS as [n][k]): lane n=lane&15,
//              elt j -> K = (j<8?0:16)+h*8+(j&7)  (same order as A)
// C/D 16x16 f32: vgpr r, value at (M = r + (lane>=16?8:0), N = lane&15)
// ---------------------------------------------------------------------------
#define LDS_PITCH 40   // bf16 elements per row (+8 pad); 80B = multiple of 16B

__device__ __forceinline__ v16bf load_a_frag(const bf16_t* rowPtr, int h) {
    v8bf lo = *(const v8bf*)(rowPtr + h * 8);        // K = h*8 .. h*8+7
    v8bf hi = *(const v8bf*)(rowPtr + 16 + h * 8);   // K = 16+h*8 .. 16+h*8+7
    return __builtin_shufflevector(lo, hi, 0,1,2,3,4,5,6,7,8,9,10,11,12,13,14,15);
}

__device__ __forceinline__ v16bf load_b_frag(const bf16_t* rowPtr, int h) {
    v8bf lo = *(const v8bf*)(rowPtr + h * 8);
    v8bf hi = *(const v8bf*)(rowPtr + 16 + h * 8);
    return __builtin_shufflevector(lo, hi, 0,1,2,3,4,5,6,7,8,9,10,11,12,13,14,15);
}

// ---------------------------------------------------------------------------
// sims GEMM: sims[m][n] = (mag[m,:] . cmag[n,:]) * inv_mnorm[m] * inv_cnorm[n]
// A: magbf [MROWS][DDIM] bf16, B: cmagT [DDIM][NPAD] bf16 (already K-major)
// CTA tile 128x128, BK=32, 8 waves each 32x64 (2x4 WMMA accumulators)
// ---------------------------------------------------------------------------
__global__ void __launch_bounds__(256, 1)
gemm_sims_kernel(const bf16_t* __restrict__ A, const bf16_t* __restrict__ Bt,
                 const float* __restrict__ inv_m, const float* __restrict__ inv_c,
                 float* __restrict__ sims)
{
    __shared__ __align__(16) bf16_t sA[128][LDS_PITCH];
    __shared__ __align__(16) bf16_t sB[128][LDS_PITCH];   // sB[n][k]

    const int tid  = threadIdx.x;
    const int lane = tid & 31;
    const int wave = tid >> 5;
    const int m0 = blockIdx.x * 128;
    const int n0 = blockIdx.y * 128;
    const int wm0 = (wave & 3) * 32;
    const int wn0 = (wave >> 2) * 64;
    const int h   = lane >> 4;
    const int l15 = lane & 15;

    // A staging: 512 16B-chunks, 2 per thread (async global->LDS)
    const int ar0 = tid >> 2, ao0 = (tid & 3) * 8;
    const uint32_t ldsA0 = (uint32_t)(uintptr_t)&sA[ar0][ao0];
    const uint32_t ldsA1 = (uint32_t)(uintptr_t)&sA[ar0 + 64][ao0];
    // B staging: k-row pairs; unit = (pr, 8-col chunk); 256 units, 1 per thread
    const int bpr = tid >> 4, bc8 = (tid & 15) * 8;

    v8f acc[2][4] = {};

    for (int k0 = 0; k0 < DDIM; k0 += 32) {
        // ---- A tile (128x32 bf16): async DMA into LDS ----
        async_copy_b128(ldsA0, &A[(size_t)(m0 + ar0) * DDIM + k0 + ao0]);
        async_copy_b128(ldsA1, &A[(size_t)(m0 + ar0 + 64) * DDIM + k0 + ao0]);
        // ---- B tile (32x128 bf16), transpose into sB[n][k], packed k-pairs ----
        {
            v8bf r0 = *(const v8bf*)&Bt[(size_t)(k0 + 2 * bpr)     * NPAD + n0 + bc8];
            v8bf r1 = *(const v8bf*)&Bt[(size_t)(k0 + 2 * bpr + 1) * NPAD + n0 + bc8];
#pragma unroll
            for (int j = 0; j < 8; ++j) {
                v2bf p; p[0] = r0[j]; p[1] = r1[j];
                *(v2bf*)&sB[bc8 + j][2 * bpr] = p;
            }
        }
        if (k0 + 32 < DDIM)
            __builtin_prefetch(&Bt[(size_t)(k0 + 32 + 2 * bpr) * NPAD + n0 + bc8], 0, 0);
        wait_async0();
        __syncthreads();

        v16bf af0 = load_a_frag(&sA[wm0 + l15][0], h);
        v16bf af1 = load_a_frag(&sA[wm0 + 16 + l15][0], h);
#pragma unroll
        for (int ni = 0; ni < 4; ++ni) {
            v16bf bfr = load_b_frag(&sB[wn0 + ni * 16 + l15][0], h);
            acc[0][ni] = __builtin_amdgcn_wmma_f32_16x16x32_bf16(
                false, af0, false, bfr, (short)0, acc[0][ni], false, false);
            acc[1][ni] = __builtin_amdgcn_wmma_f32_16x16x32_bf16(
                false, af1, false, bfr, (short)0, acc[1][ni], false, false);
        }
        __syncthreads();
    }

#pragma unroll
    for (int mi = 0; mi < 2; ++mi)
#pragma unroll
        for (int ni = 0; ni < 4; ++ni)
#pragma unroll
            for (int r = 0; r < 8; ++r) {
                int row = m0 + wm0 + mi * 16 + r + (h << 3);
                int col = n0 + wn0 + ni * 16 + l15;
                sims[(size_t)row * NPAD + col] = acc[mi][ni][r] * inv_m[row] * inv_c[col];
            }
}

// ---------------------------------------------------------------------------
// MLP GEMM: C = act(A @ B + bias). A bf16 [MROWS][K] (workspace),
// B f32 [K][N] (input weights, converted to bf16 while staging into LDS).
// ---------------------------------------------------------------------------
template <bool GELU, bool OUT_BF16>
__global__ void __launch_bounds__(256, 1)
gemm_mlp_kernel(const bf16_t* __restrict__ A, const float* __restrict__ B,
                const float* __restrict__ bias, void* __restrict__ Cout,
                int K, int N)
{
    __shared__ __align__(16) bf16_t sA[128][LDS_PITCH];
    __shared__ __align__(16) bf16_t sB[128][LDS_PITCH];   // sB[n][k]

    const int tid  = threadIdx.x;
    const int lane = tid & 31;
    const int wave = tid >> 5;
    const int m0 = blockIdx.x * 128;
    const int n0 = blockIdx.y * 128;
    const int wm0 = (wave & 3) * 32;
    const int wn0 = (wave >> 2) * 64;
    const int h   = lane >> 4;
    const int l15 = lane & 15;

    const int ar0 = tid >> 2, ao0 = (tid & 3) * 8;          // A: 16B chunks
    const uint32_t ldsA0 = (uint32_t)(uintptr_t)&sA[ar0][ao0];
    const uint32_t ldsA1 = (uint32_t)(uintptr_t)&sA[ar0 + 64][ao0];
    // B: k-row-pair x float4 units: 16 * 32 = 512 units, 2 per thread
    const int bpr = tid >> 5, bc4 = (tid & 31) * 4;

    v8f acc[2][4] = {};

    for (int k0 = 0; k0 < K; k0 += 32) {
        // ---- A tile (128x32 bf16): async DMA into LDS ----
        async_copy_b128(ldsA0, &A[(size_t)(m0 + ar0) * K + k0 + ao0]);
        async_copy_b128(ldsA1, &A[(size_t)(m0 + ar0 + 64) * K + k0 + ao0]);
        // ---- B tile (32x128 f32 -> bf16), transposed, packed k-pairs ----
#pragma unroll
        for (int it = 0; it < 2; ++it) {
            int pr  = bpr + it * 8;          // k-pair index 0..15
            int col = n0 + bc4;
            float4 r0 = make_float4(0.f, 0.f, 0.f, 0.f);
            float4 r1 = make_float4(0.f, 0.f, 0.f, 0.f);
            if (col < N) {                   // N % 4 == 0, chunk fully in/out
                r0 = *(const float4*)&B[(size_t)(k0 + 2 * pr)     * N + col];
                r1 = *(const float4*)&B[(size_t)(k0 + 2 * pr + 1) * N + col];
            }
            v2bf p;
            p[0] = (bf16_t)r0.x; p[1] = (bf16_t)r1.x; *(v2bf*)&sB[bc4 + 0][2 * pr] = p;
            p[0] = (bf16_t)r0.y; p[1] = (bf16_t)r1.y; *(v2bf*)&sB[bc4 + 1][2 * pr] = p;
            p[0] = (bf16_t)r0.z; p[1] = (bf16_t)r1.z; *(v2bf*)&sB[bc4 + 2][2 * pr] = p;
            p[0] = (bf16_t)r0.w; p[1] = (bf16_t)r1.w; *(v2bf*)&sB[bc4 + 3][2 * pr] = p;
        }
        if (k0 + 32 < K && n0 + bc4 < N)
            __builtin_prefetch(&B[(size_t)(k0 + 32 + 2 * bpr) * N + n0 + bc4], 0, 0);
        wait_async0();
        __syncthreads();

        v16bf af0 = load_a_frag(&sA[wm0 + l15][0], h);
        v16bf af1 = load_a_frag(&sA[wm0 + 16 + l15][0], h);
#pragma unroll
        for (int ni = 0; ni < 4; ++ni) {
            v16bf bfr = load_b_frag(&sB[wn0 + ni * 16 + l15][0], h);
            acc[0][ni] = __builtin_amdgcn_wmma_f32_16x16x32_bf16(
                false, af0, false, bfr, (short)0, acc[0][ni], false, false);
            acc[1][ni] = __builtin_amdgcn_wmma_f32_16x16x32_bf16(
                false, af1, false, bfr, (short)0, acc[1][ni], false, false);
        }
        __syncthreads();
    }

#pragma unroll
    for (int mi = 0; mi < 2; ++mi)
#pragma unroll
        for (int ni = 0; ni < 4; ++ni) {
            int col = n0 + wn0 + ni * 16 + l15;
            if (col >= N) continue;
            float bv = bias[col];
#pragma unroll
            for (int r = 0; r < 8; ++r) {
                int row = m0 + wm0 + mi * 16 + r + (h << 3);
                float v = acc[mi][ni][r] + bv;
                if (GELU) v = gelu_exact(v);
                if (OUT_BF16)
                    ((bf16_t*)Cout)[(size_t)row * N + col] = (bf16_t)v;
                else
                    ((float*)Cout)[(size_t)row * N + col] = v;
            }
        }
}

// ---------------------------------------------------------------------------
// top-k (k=5) per row over NCON sims; descending, lower-index tie-break
// ---------------------------------------------------------------------------
__global__ void topk_kernel(const float* __restrict__ sims,
                            float* __restrict__ top_sims, int* __restrict__ top_idx)
{
    __shared__ float vals[NPAD];
    __shared__ float rmax[256];
    __shared__ int   ridx[256];
    const int row = blockIdx.x;
    const int tid = threadIdx.x;

    for (int i = tid; i < NCON; i += 256)
        vals[i] = sims[(size_t)row * NPAD + i];
    __syncthreads();

    for (int t = 0; t < TOPK; ++t) {
        float best = -3.4e38f;
        int bi = 0;
        for (int i = tid; i < NCON; i += 256) {
            float v = vals[i];
            if (v > best) { best = v; bi = i; }
        }
        rmax[tid] = best;
        ridx[tid] = bi;
        __syncthreads();
        for (int st = 128; st > 0; st >>= 1) {
            if (tid < st) {
                float a = rmax[tid], b = rmax[tid + st];
                if (b > a || (b == a && ridx[tid + st] < ridx[tid])) {
                    rmax[tid] = b;
                    ridx[tid] = ridx[tid + st];
                }
            }
            __syncthreads();
        }
        if (tid == 0) {
            top_sims[(size_t)row * TOPK + t] = rmax[0];
            top_idx [(size_t)row * TOPK + t] = ridx[0];
            vals[ridx[0]] = -3.4e38f;
        }
        __syncthreads();
    }
}

// ---------------------------------------------------------------------------
extern "C" void kernel_launch(void* const* d_in, const int* in_sizes, int n_in,
                              void* d_out, int out_size, void* d_ws, size_t ws_size,
                              hipStream_t stream)
{
    const float* sr = (const float*)d_in[0];
    const float* si = (const float*)d_in[1];
    const float* cr = (const float*)d_in[2];
    const float* ci = (const float*)d_in[3];
    const float* W1 = (const float*)d_in[4];
    const float* b1 = (const float*)d_in[5];
    const float* W2 = (const float*)d_in[6];
    const float* b2 = (const float*)d_in[7];
    const float* W3 = (const float*)d_in[8];
    const float* b3 = (const float*)d_in[9];
    const float* W4 = (const float*)d_in[10];
    const float* b4 = (const float*)d_in[11];
    // d_in[12] = top_k (fixed at 5 for this problem instance)

    // workspace layout (~28 MB)
    char* ws = (char*)d_ws;
    bf16_t* magbf     = (bf16_t*)ws;  ws += (size_t)MROWS * DDIM * sizeof(bf16_t);
    float*  inv_mnorm = (float*)ws;   ws += (size_t)MROWS * sizeof(float);
    bf16_t* cmagT     = (bf16_t*)ws;  ws += (size_t)DDIM * NPAD * sizeof(bf16_t);
    float*  inv_cnorm = (float*)ws;   ws += (size_t)NPAD * sizeof(float);
    bf16_t* xcat      = (bf16_t*)ws;  ws += (size_t)MROWS * 2 * DDIM * sizeof(bf16_t);
    float*  sims      = (float*)ws;   ws += (size_t)MROWS * NPAD * sizeof(float);
    bf16_t* h1        = (bf16_t*)ws;  ws += (size_t)MROWS * DDIM * sizeof(bf16_t);
    bf16_t* h2        = (bf16_t*)ws;  ws += (size_t)MROWS * 512 * sizeof(bf16_t);
    bf16_t* h3        = (bf16_t*)ws;  ws += (size_t)MROWS * DDIM * sizeof(bf16_t);

    float* top_sims = (float*)d_out;
    int*   top_idx  = (int*)d_out + (size_t)MROWS * TOPK;
    float* logits   = (float*)d_out + (size_t)2 * MROWS * TOPK;

    prep_mag_kernel<<<MROWS, 256, 0, stream>>>(sr, si, magbf, inv_mnorm, xcat);
    prep_concept_kernel<<<NPAD, 256, 0, stream>>>(cr, ci, cmagT, inv_cnorm);

    dim3 gs(MROWS / 128, NPAD / 128);
    gemm_sims_kernel<<<gs, 256, 0, stream>>>(magbf, cmagT, inv_mnorm, inv_cnorm, sims);

    topk_kernel<<<MROWS, 256, 0, stream>>>(sims, top_sims, top_idx);

    dim3 g1(MROWS / 128, DDIM / 128);
    gemm_mlp_kernel<true,  true ><<<g1, 256, 0, stream>>>(xcat, W1, b1, h1, 2 * DDIM, DDIM);
    dim3 g2(MROWS / 128, 512 / 128);
    gemm_mlp_kernel<true,  true ><<<g2, 256, 0, stream>>>(h1, W2, b2, h2, DDIM, 512);
    dim3 g3(MROWS / 128, DDIM / 128);
    gemm_mlp_kernel<true,  true ><<<g3, 256, 0, stream>>>(h2, W3, b3, h3, 512, DDIM);
    dim3 g4(MROWS / 128, (VOUT + 127) / 128);
    gemm_mlp_kernel<false, false><<<g4, 256, 0, stream>>>(h3, W4, b4, logits, DDIM, VOUT);
}
